// LoRALinearExpertsWrapper_86242943303913
// MI455X (gfx1250) — matrix operand order repaired and checked by reading, not compiled, not run on hardware
//
#include <hip/hip_runtime.h>
#include <cstdint>
#include <cstddef>

// ---------------- problem constants ----------------
#define K_E 8
#define K_T 2048
#define K_H 2048
#define K_D 4096
#define K_R 8
#define K_SCALE 2.0f

// ---------------- CDNA5 WMMA types ----------------
typedef __attribute__((ext_vector_type(4)))  unsigned int u32x4;
typedef __attribute__((ext_vector_type(8)))  float        v8f;
typedef __attribute__((ext_vector_type(16))) __bf16       v16bf;

union Frag { v16bf v; u32x4 q[2]; };

__device__ __forceinline__ unsigned short f2bf_bits(float f) {
  unsigned u = __builtin_bit_cast(unsigned, f);
  u += 0x7FFFu + ((u >> 16) & 1u);           // round-to-nearest-even
  return (unsigned short)(u >> 16);
}
__device__ __forceinline__ unsigned int pk_bf2(float lo, float hi) {
  return (unsigned)f2bf_bits(lo) | ((unsigned)f2bf_bits(hi) << 16);
}
__device__ __forceinline__ float bf_to_f(unsigned short b) {
  return __builtin_bit_cast(float, ((unsigned)b) << 16);
}

// LDS byte-offset of a __shared__ pointer (generic ptr low 32 bits = LDS offset)
__device__ __forceinline__ unsigned lds_off(const void* p) {
  return (unsigned)(unsigned long long)(uintptr_t)p;
}

// CDNA5 async DMA: 16B global -> LDS, tracked with ASYNCcnt (ISA 10. / §15.18.3 op 98)
__device__ __forceinline__ void async_ld16(unsigned ldsoff, const void* sbase, unsigned voff) {
  asm volatile("global_load_async_to_lds_b128 %0, %1, %2 offset:0"
               :: "v"(ldsoff), "v"(voff), "s"(sbase)
               : "memory");
}
__device__ __forceinline__ void wait_async0() {
  asm volatile("s_wait_asynccnt 0x0" ::: "memory");
}

// =====================================================================
// fp32 -> bf16 bulk conversion (memory-bound, done ONCE per tensor)
// =====================================================================
__global__ __launch_bounds__(256) void k_cvt_bf16(const float* __restrict__ src,
                                                  unsigned short* __restrict__ dst,
                                                  size_t n4) {
  size_t i = (size_t)blockIdx.x * 256 + threadIdx.x;
  const size_t stride = (size_t)gridDim.x * 256;
  for (; i < n4; i += stride) {
    float4 v = ((const float4*)src)[i];
    uint2 o; o.x = pk_bf2(v.x, v.y); o.y = pk_bf2(v.z, v.w);
    ((uint2*)dst)[i] = o;
  }
}

// =====================================================================
// Rank-8 LoRA projection: low[row][r] = sum_k Xbf16[row][k] * A[e][r][k]
// =====================================================================
__global__ __launch_bounds__(256) void k_low_bf16(const unsigned short* __restrict__ X,
                                                  const float* __restrict__ A,
                                                  float* __restrict__ low, int Kdim) {
  const int row = blockIdx.x;
  const int e   = row / K_T;
  const unsigned short* x = X + (size_t)row * Kdim;
  const float* a = A + (size_t)e * K_R * Kdim;
  float p[K_R];
#pragma unroll
  for (int r = 0; r < K_R; ++r) p[r] = 0.f;
  for (int k = threadIdx.x; k < Kdim; k += 256) {
    float xv = bf_to_f(x[k]);
#pragma unroll
    for (int r = 0; r < K_R; ++r) p[r] += xv * a[(size_t)r * Kdim + k];
  }
  __shared__ float red[256];
  for (int r = 0; r < K_R; ++r) {
    red[threadIdx.x] = p[r];
    __syncthreads();
    for (int s = 128; s > 0; s >>= 1) {
      if ((int)threadIdx.x < s) red[threadIdx.x] += red[threadIdx.x + s];
      __syncthreads();
    }
    if (threadIdx.x == 0) low[(size_t)row * K_R + r] = red[0];
    __syncthreads();
  }
}

// =====================================================================
// GEMM1 (fused gate+up + LoRA + SwiGLU), all-bf16 operands.
// Block: 128(M) x 64 cols of D, computes BOTH gate & up for those cols.
// Tile feed: global_load_async_to_lds_b128 (ASYNCcnt), double-buffered.
// 8 waves, wave = 32x32 gate + 32x32 up => 8 WMMA / K-step.
// =====================================================================
__global__ __launch_bounds__(256) void k_gemm1(const unsigned short* __restrict__ Xb,
                                               const unsigned short* __restrict__ Wb,
                                               const float* __restrict__ Bgu,
                                               const float* __restrict__ lowg,
                                               unsigned short* __restrict__ Hbuf) {
  __shared__ __align__(16) unsigned short As[2][128][40];
  __shared__ __align__(16) unsigned short Bg[2][64][40];
  __shared__ __align__(16) unsigned short Bu[2][64][40];

  const int e  = blockIdx.z;
  const int mb = blockIdx.y;      // T/128
  const int jb = blockIdx.x;      // D/64
  const int tid  = threadIdx.x;
  const int lane = tid & 31, wid = tid >> 5;
  const int wm = wid >> 1, wn = wid & 1;        // 4 x 2 wave grid
  const int hi16 = lane >> 4, ln = lane & 15;

  const unsigned short* xg    = Xb + ((size_t)e * K_T + (size_t)mb * 128) * K_H;
  const unsigned short* wgate = Wb + ((size_t)e * 2 * K_D + (size_t)jb * 64) * K_H;
  const unsigned short* wup   = Wb + ((size_t)e * 2 * K_D + K_D + (size_t)jb * 64) * K_H;

  const unsigned asb = lds_off(&As[0][0][0]);
  const unsigned bgb = lds_off(&Bg[0][0][0]);
  const unsigned bub = lds_off(&Bu[0][0][0]);

  // 16B chunk per lane: chunk row = tid>>2, chunk col = (tid&3)*8 bf16
  const unsigned crow = (unsigned)tid >> 2;     // 0..63
  const unsigned ccol = ((unsigned)tid & 3u) * 8u;

  auto issue = [&](int buf, int k0) {
    const unsigned b = (unsigned)buf;
    async_ld16(asb + ((b * 128 + crow) * 40 + ccol) * 2,      xg,
               (crow * K_H + (unsigned)k0 + ccol) * 2);
    async_ld16(asb + ((b * 128 + crow + 64) * 40 + ccol) * 2, xg,
               ((crow + 64) * K_H + (unsigned)k0 + ccol) * 2);
    async_ld16(bgb + ((b * 64 + crow) * 40 + ccol) * 2,       wgate,
               (crow * K_H + (unsigned)k0 + ccol) * 2);
    async_ld16(bub + ((b * 64 + crow) * 40 + ccol) * 2,       wup,
               (crow * K_H + (unsigned)k0 + ccol) * 2);
  };

  v8f accg[2][2], accu[2][2];
#pragma unroll
  for (int i = 0; i < 2; ++i)
#pragma unroll
    for (int j = 0; j < 2; ++j) { accg[i][j] = v8f{}; accu[i][j] = v8f{}; }

  issue(0, 0);

  const int nk = K_H / 32;   // 64
  for (int kt = 0; kt < nk; ++kt) {
    const int cur = kt & 1;
    wait_async0();           // our 4 chunks for buf 'cur' have landed
    __syncthreads();         // everyone's chunks landed; prev reads of buf cur^1 done
    if (kt + 1 < nk) issue(cur ^ 1, (kt + 1) * 32);   // DMA next tile during WMMA

    Frag a[2], fg[2], fu[2];
#pragma unroll
    for (int mi = 0; mi < 2; ++mi) {
      const unsigned short* pr = &As[cur][wm * 32 + mi * 16 + ln][hi16 * 16];
      a[mi].q[0] = *(const u32x4*)pr;
      a[mi].q[1] = *(const u32x4*)(pr + 8);
    }
#pragma unroll
    for (int ni = 0; ni < 2; ++ni) {
      const unsigned short* pg = &Bg[cur][wn * 32 + ni * 16 + ln][hi16 * 16];
      fg[ni].q[0] = *(const u32x4*)pg;
      fg[ni].q[1] = *(const u32x4*)(pg + 8);
      const unsigned short* pu = &Bu[cur][wn * 32 + ni * 16 + ln][hi16 * 16];
      fu[ni].q[0] = *(const u32x4*)pu;
      fu[ni].q[1] = *(const u32x4*)(pu + 8);
    }
#pragma unroll
    for (int mi = 0; mi < 2; ++mi)
#pragma unroll
      for (int ni = 0; ni < 2; ++ni) {
        accg[mi][ni] = __builtin_amdgcn_wmma_f32_16x16x32_bf16(
            false, a[mi].v, false, fg[ni].v, (short)0, accg[mi][ni], false, false);
        accu[mi][ni] = __builtin_amdgcn_wmma_f32_16x16x32_bf16(
            false, a[mi].v, false, fu[ni].v, (short)0, accu[mi][ni], false, false);
      }
  }

  // epilogue: + LoRA deltas, SwiGLU, write h as bf16
  const int mbase = mb * 128 + wm * 32;
  const int nbase = jb * 64 + wn * 32;
#pragma unroll
  for (int ni = 0; ni < 2; ++ni) {
    const int ncol = nbase + ni * 16 + ln;                 // column within D
    const float* bgr = Bgu + ((size_t)e * 2 * K_D + ncol) * K_R;
    const float* bur = Bgu + ((size_t)e * 2 * K_D + K_D + ncol) * K_R;
    float4 g0 = *(const float4*)bgr, g1 = *(const float4*)(bgr + 4);
    float4 u0 = *(const float4*)bur, u1 = *(const float4*)(bur + 4);
#pragma unroll
    for (int mi = 0; mi < 2; ++mi) {
#pragma unroll
      for (int r = 0; r < 8; ++r) {
        const int mrow = mbase + mi * 16 + hi16 * 8 + r;   // C/D layout: M = r + 8*half
        const float* lw = lowg + ((size_t)e * K_T + mrow) * K_R;
        float4 l0 = *(const float4*)lw, l1 = *(const float4*)(lw + 4);
        float dg = l0.x * g0.x + l0.y * g0.y + l0.z * g0.z + l0.w * g0.w +
                   l1.x * g1.x + l1.y * g1.y + l1.z * g1.z + l1.w * g1.w;
        float du = l0.x * u0.x + l0.y * u0.y + l0.z * u0.z + l0.w * u0.w +
                   l1.x * u1.x + l1.y * u1.y + l1.z * u1.z + l1.w * u1.w;
        float gv = accg[mi][ni][r] + K_SCALE * dg;
        float uv = accu[mi][ni][r] + K_SCALE * du;
        float hv = uv * (gv / (1.f + __expf(-gv)));        // up * silu(gate)
        Hbuf[((size_t)e * K_T + mrow) * K_D + ncol] = f2bf_bits(hv);
      }
    }
  }
}

// =====================================================================
// GEMM2 (down-proj + LoRA), all-bf16 operands, async LDS feed.
// Block: 128(M) x 128(N=H cols); 8 waves of 32x64 (2x4 WMMA tiles).
// =====================================================================
__global__ __launch_bounds__(256) void k_gemm2(const unsigned short* __restrict__ Hb,
                                               const unsigned short* __restrict__ Wb,
                                               const float* __restrict__ Bd,
                                               const float* __restrict__ lowd,
                                               float* __restrict__ Out) {
  __shared__ __align__(16) unsigned short As[2][128][40];
  __shared__ __align__(16) unsigned short Bs[2][128][40];

  const int e  = blockIdx.z;
  const int mb = blockIdx.y;   // T/128
  const int nb = blockIdx.x;   // H/128
  const int tid  = threadIdx.x;
  const int lane = tid & 31, wid = tid >> 5;
  const int wm = wid >> 1, wn = wid & 1;
  const int hi16 = lane >> 4, ln = lane & 15;

  const unsigned short* hg = Hb + ((size_t)e * K_T + (size_t)mb * 128) * K_D;
  const unsigned short* wr = Wb + ((size_t)e * K_H + (size_t)nb * 128) * K_D;

  const unsigned asb = lds_off(&As[0][0][0]);
  const unsigned bsb = lds_off(&Bs[0][0][0]);

  const unsigned crow = (unsigned)tid >> 2;     // 0..63
  const unsigned ccol = ((unsigned)tid & 3u) * 8u;

  auto issue = [&](int buf, int k0) {
    const unsigned b = (unsigned)buf;
    async_ld16(asb + ((b * 128 + crow) * 40 + ccol) * 2,      hg,
               (crow * K_D + (unsigned)k0 + ccol) * 2);
    async_ld16(asb + ((b * 128 + crow + 64) * 40 + ccol) * 2, hg,
               ((crow + 64) * K_D + (unsigned)k0 + ccol) * 2);
    async_ld16(bsb + ((b * 128 + crow) * 40 + ccol) * 2,      wr,
               (crow * K_D + (unsigned)k0 + ccol) * 2);
    async_ld16(bsb + ((b * 128 + crow + 64) * 40 + ccol) * 2, wr,
               ((crow + 64) * K_D + (unsigned)k0 + ccol) * 2);
  };

  v8f acc[2][4];
#pragma unroll
  for (int i = 0; i < 2; ++i)
#pragma unroll
    for (int j = 0; j < 4; ++j) acc[i][j] = v8f{};

  issue(0, 0);

  const int nk = K_D / 32;   // 128
  for (int kt = 0; kt < nk; ++kt) {
    const int cur = kt & 1;
    wait_async0();
    __syncthreads();
    if (kt + 1 < nk) issue(cur ^ 1, (kt + 1) * 32);

    Frag a[2], b[4];
#pragma unroll
    for (int mi = 0; mi < 2; ++mi) {
      const unsigned short* pr = &As[cur][wm * 32 + mi * 16 + ln][hi16 * 16];
      a[mi].q[0] = *(const u32x4*)pr;
      a[mi].q[1] = *(const u32x4*)(pr + 8);
    }
#pragma unroll
    for (int ni = 0; ni < 4; ++ni) {
      const unsigned short* pb = &Bs[cur][wn * 64 + ni * 16 + ln][hi16 * 16];
      b[ni].q[0] = *(const u32x4*)pb;
      b[ni].q[1] = *(const u32x4*)(pb + 8);
    }
#pragma unroll
    for (int mi = 0; mi < 2; ++mi)
#pragma unroll
      for (int ni = 0; ni < 4; ++ni)
        acc[mi][ni] = __builtin_amdgcn_wmma_f32_16x16x32_bf16(
            false, a[mi].v, false, b[ni].v, (short)0, acc[mi][ni], false, false);
  }

  const int mbase = mb * 128 + wm * 32;
  const int nbase = nb * 128 + wn * 64;
#pragma unroll
  for (int ni = 0; ni < 4; ++ni) {
    const int ncol = nbase + ni * 16 + ln;                 // column within H
    const float* bdr = Bd + ((size_t)e * K_H + ncol) * K_R;
    float4 b0 = *(const float4*)bdr, b1 = *(const float4*)(bdr + 4);
#pragma unroll
    for (int mi = 0; mi < 2; ++mi) {
#pragma unroll
      for (int r = 0; r < 8; ++r) {
        const int mrow = mbase + mi * 16 + hi16 * 8 + r;
        const float* lw = lowd + ((size_t)e * K_T + mrow) * K_R;
        float4 l0 = *(const float4*)lw, l1 = *(const float4*)(lw + 4);
        float dd = l0.x * b0.x + l0.y * b0.y + l0.z * b0.z + l0.w * b0.w +
                   l1.x * b1.x + l1.y * b1.y + l1.z * b1.z + l1.w * b1.w;
        Out[((size_t)e * K_T + mrow) * K_H + ncol] = acc[mi][ni][r] + K_SCALE * dd;
      }
    }
  }
}

// =====================================================================
extern "C" void kernel_launch(void* const* d_in, const int* in_sizes, int n_in,
                              void* d_out, int out_size, void* d_ws, size_t ws_size,
                              hipStream_t stream) {
  (void)in_sizes; (void)n_in; (void)out_size; (void)ws_size;
  const float* X    = (const float*)d_in[0];   // [E*T, H]
  const float* W_gu = (const float*)d_in[1];   // [E, 2D, H]
  const float* A_gu = (const float*)d_in[2];   // [E, R, H]
  const float* B_gu = (const float*)d_in[3];   // [E, 2D, R]
  const float* W_d  = (const float*)d_in[4];   // [E, H, D]
  const float* A_d  = (const float*)d_in[5];   // [E, R, D]
  const float* B_d  = (const float*)d_in[6];   // [E, H, R]
  float* Out = (float*)d_out;                  // [E*T, H] fp32

  const size_t nX  = (size_t)K_E * K_T * K_H;      // 33.5M
  const size_t nWg = (size_t)K_E * 2 * K_D * K_H;  // 134M
  const size_t nWd = (size_t)K_E * K_H * K_D;      // 67M
  const size_t nH  = (size_t)K_E * K_T * K_D;      // 67M

  // workspace layout (bf16 tensors + rank-8 intermediates)
  float* low_gu = (float*)d_ws;                                   // 512KB
  float* low_d  = low_gu + (size_t)K_E * K_T * K_R;               // 512KB
  unsigned short* hbuf = (unsigned short*)(low_d + (size_t)K_E * K_T * K_R); // nH bf16
  unsigned short* xb   = hbuf + nH;                               // nX bf16
  unsigned short* wgub = xb + nX;                                 // nWg bf16
  unsigned short* wdb  = wgub + nWg;                              // nWd bf16

  // 1) one-shot fp32 -> bf16 conversions (memory-bound)
  k_cvt_bf16<<<8192, 256, 0, stream>>>(X,    xb,   nX  / 4);
  k_cvt_bf16<<<8192, 256, 0, stream>>>(W_gu, wgub, nWg / 4);
  k_cvt_bf16<<<8192, 256, 0, stream>>>(W_d,  wdb,  nWd / 4);

  // 2) LoRA low-rank projection for gate/up
  k_low_bf16<<<K_E * K_T, 256, 0, stream>>>(xb, A_gu, low_gu, K_H);

  // 3) fused gate/up GEMM + LoRA + SwiGLU -> h (bf16)
  dim3 g1(K_D / 64, K_T / 128, K_E);
  k_gemm1<<<g1, 256, 0, stream>>>(xb, wgub, B_gu, low_gu, hbuf);

  // 4) LoRA low-rank projection for down
  k_low_bf16<<<K_E * K_T, 256, 0, stream>>>(hbuf, A_d, low_d, K_D);

  // 5) down GEMM + LoRA -> out (fp32)
  dim3 g2(K_H / 128, K_T / 128, K_E);
  k_gemm2<<<g2, 256, 0, stream>>>(hbuf, wdb, B_d, low_d, Out);
}
